// ShiftNMF_78606491452305
// MI455X (gfx1250) — matrix operand order repaired and checked by reading, not compiled, not run on hardware
//
#include <hip/hip_runtime.h>
#include <hip/hip_fp16.h>

typedef __attribute__((ext_vector_type(16))) _Float16 v16h;
typedef __attribute__((ext_vector_type(8)))  float    v8f;

#define NROW 1024
#define RANK 8
#define MLEN 4096
#define PI_F 3.14159265358979323846f

__device__ __forceinline__ float softplus_f(float x) {
    // numerically stable softplus
    return (x > 20.0f) ? x : log1pf(__expf(x));
}

// ---------------------------------------------------------------------------
// Kernel 0: Hp16[d][t] = (f16) softplus(H[d][t])
// ---------------------------------------------------------------------------
__global__ __launch_bounds__(256) void prep_softplus_h(
        const float* __restrict__ H, _Float16* __restrict__ Hp16) {
    int i = blockIdx.x * blockDim.x + threadIdx.x;
    if (i < RANK * MLEN) Hp16[i] = (_Float16)softplus_f(H[i]);
}

// ---------------------------------------------------------------------------
// Kernel 1: DFT as GEMM via v_wmma_f32_16x16x32_f16.
//   Hft[d][m] = sum_t Hp[d][t] * exp(-2*pi*i * t*m / M)
// One wave (32 threads) per 16-column m-tile; K swept 32 at a time.
// A (16x32 f16): rows = d (8 real rows, 8 zero-pad), contiguous K order:
//   lane L holds row M=L&15; half h -> K = (L>=16?16:0) + h
//   (same K order as B).
// B (32x16 f16): lane L, half h -> K = (L>=16?16:0)+h, N = L&15
//   (K split across lane halves, contiguous within a lane — this code's
//    own order, not AMD's table order).
// Twiddles advanced by a complex rotator (no per-step sincos).
// Result stored transposed: Hft_t[m][d] (float2), 64B per m.
// ---------------------------------------------------------------------------
__global__ __launch_bounds__(32) void dft_wmma(
        const _Float16* __restrict__ Hp16, float2* __restrict__ Hft_t) {
    const int lane = threadIdx.x;          // wave32
    const int m0   = blockIdx.x * 16;
    const int col  = m0 + (lane & 15);     // m column this lane supplies (B)
    const int row  = lane & 15;            // d row this lane supplies (A)
    const int kHi  = (lane >= 16) ? 16 : 0;

    // tw[h] = exp(-2*pi*i * (kHi+h) * col / M), advanced by R = exp(-2*pi*i*32*col/M)
    const float wm = -2.0f * PI_F * (float)col / (float)MLEN;
    float twc[16], tws[16];
#pragma unroll
    for (int h = 0; h < 16; ++h) {
        __sincosf(wm * (float)(kHi + h), &tws[h], &twc[h]);
    }
    float rs, rc;
    __sincosf(wm * 32.0f, &rs, &rc);

    v8f acc_re = {};
    v8f acc_im = {};

    for (int t0 = 0; t0 < MLEN; t0 += 32) {
        // ---- A fragment (softplus(H) rows, f16) ----
        v16h a;
#pragma unroll
        for (int h = 0; h < 16; ++h) {
            const int i  = h >> 1, j = h & 1;
            const int grp = i >> 2, wi = i & 3;
            const int K  = ((lane >= 16) ? 16 : 0) + h; (void)grp; (void)wi; (void)j;
            a[h] = (row < RANK) ? Hp16[row * MLEN + t0 + K] : (_Float16)0.0f;
        }
        // ---- B fragments: cos tile and (-sin) tile ----
        v16h bc, bs;
#pragma unroll
        for (int h = 0; h < 16; ++h) {
            bc[h] = (_Float16)twc[h];
            bs[h] = (_Float16)(-tws[h]);   // Im(Hft) = -sum Hp*sin
        }
        acc_re = __builtin_amdgcn_wmma_f32_16x16x32_f16(
                     false, a, false, bc, (short)0, acc_re, false, false);
        acc_im = __builtin_amdgcn_wmma_f32_16x16x32_f16(
                     false, a, false, bs, (short)0, acc_im, false, false);
        // ---- rotate twiddles forward by 32 in t ----
#pragma unroll
        for (int h = 0; h < 16; ++h) {
            const float c = twc[h] * rc - tws[h] * rs;
            const float s = tws[h] * rc + twc[h] * rs;
            twc[h] = c; tws[h] = s;
        }
    }

    // C/D layout: VGPR r, lanes 0-15 -> (M=r, N=lane); lanes 16-31 -> M=8+r
    // (rows 8..15 are zero padding -> discard). Store transposed.
    if (lane < 16) {
#pragma unroll
        for (int r = 0; r < 8; ++r) {
            Hft_t[(size_t)(m0 + lane) * RANK + r] =
                make_float2(acc_re[r], acc_im[r]);
        }
    }
}

// ---------------------------------------------------------------------------
// Kernel 2: V[n][m] = sum_d softplus(W)[n][d] * exp(-2*pi*i*tau[n][d]*m/M) * Hft[d][m]
// One thread: one n, CHUNK consecutive m. Phase via geometric recurrence:
// 2 sincos per rank per CHUNK outputs, then complex rotate + complex FMA.
// Bandwidth-bound on the 33.5 MB output store.
// ---------------------------------------------------------------------------
#define CHUNK 8
__global__ __launch_bounds__(256) void shift_synth(
        const float*  __restrict__ W,
        const float*  __restrict__ tau,
        const float2* __restrict__ Hft_t,
        float2*       __restrict__ V) {
    const int n  = blockIdx.y;                                   // uniform -> SMEM loads
    const int m0 = (blockIdx.x * blockDim.x + threadIdx.x) * CHUNK;

    float wp[RANK], c[RANK], s[RANK], dc[RANK], ds[RANK];
#pragma unroll
    for (int d = 0; d < RANK; ++d) {
        wp[d] = softplus_f(W[n * RANK + d]);
        const float t = tau[n * RANK + d];
        // step rotator: exp(-2*pi*i * t / M)
        __sincosf(-2.0f * PI_F * t / (float)MLEN, &ds[d], &dc[d]);
        // base phase at m0, range-reduced in cycles
        float x = t * (float)m0 * (1.0f / (float)MLEN);
        x -= truncf(x);
        __sincosf(-2.0f * PI_F * x, &s[d], &c[d]);
    }

    const float2* hp  = Hft_t + (size_t)m0 * RANK;   // 64B-contiguous per m
    float2*       out = V + (size_t)n * MLEN + m0;

#pragma unroll
    for (int j = 0; j < CHUNK; ++j) {
        float re = 0.0f, im = 0.0f;
#pragma unroll
        for (int d = 0; d < RANK; ++d) {
            const float2 h = hp[j * RANK + d];
            // (c + i*s) already equals omega = exp(-i*2*pi*tau*m/M)
            re += wp[d] * (c[d] * h.x - s[d] * h.y);
            im += wp[d] * (c[d] * h.y + s[d] * h.x);
            // advance phase: omega *= exp(-i*2*pi*tau/M)
            const float cn = c[d] * dc[d] - s[d] * ds[d];
            const float sn = s[d] * dc[d] + c[d] * ds[d];
            c[d] = cn; s[d] = sn;
        }
        out[j] = make_float2(re, im);
    }
}

// ---------------------------------------------------------------------------
extern "C" void kernel_launch(void* const* d_in, const int* in_sizes, int n_in,
                              void* d_out, int out_size, void* d_ws, size_t ws_size,
                              hipStream_t stream) {
    (void)in_sizes; (void)n_in; (void)out_size; (void)ws_size;
    const float* W   = (const float*)d_in[0];   // [N, rank]
    const float* H   = (const float*)d_in[1];   // [rank, M]
    const float* tau = (const float*)d_in[2];   // [N, rank]

    _Float16* Hp16  = (_Float16*)d_ws;                         // 8*4096*2B  = 64 KB
    float2*   Hft_t = (float2*)((char*)d_ws + 64 * 1024);      // 4096*8*8B  = 256 KB
    float2*   V     = (float2*)d_out;                          // [N, M] complex64

    prep_softplus_h<<<dim3((RANK * MLEN + 255) / 256), dim3(256), 0, stream>>>(H, Hp16);
    dft_wmma<<<dim3(MLEN / 16), dim3(32), 0, stream>>>(Hp16, Hft_t);
    shift_synth<<<dim3(MLEN / (256 * CHUNK), NROW), dim3(256), 0, stream>>>(W, tau, Hft_t, V);
}
